// Corr1D_32392643347166
// MI455X (gfx1250) — matrix operand order repaired and checked by reading, not compiled, not run on hardware
//
#include <hip/hip_runtime.h>
#include <hip/hip_bf16.h>

// ---------------------------------------------------------------------------
// Corr1D on MI455X (gfx1250): depthwise sliding-window correlation as a
// per-(b,c) Toeplitz GEMM on v_wmma_f32_16x16x32_f16.
//
//   out[b,l,c] = inv1[b,c]*inv2[b,c] * sum_t A[b,t+l-64,c]*V[b,t,c],  l=0..128
//
// Toeplitz tile (per (b,c) pair, f16 WMMA, K=32 per step, 260 steps):
//   Afrag[m,k] = a_pad[t0 + k + m]           (a_pad[i] = a[i-64], zero halo)
//   Bfrag[k,n] = b_pad[t0 + k - 16n + 240]   (b_pad[i] = v[i-240], zero halo)
//   D[m,n]    += sum_k Afrag*Bfrag  ==> lag = m + 16n   (129 of 256 cells used)
//
// Roofline: useful 4.3 GFLOP / 134 MB in (32 FLOP/B) -> compute-bound on VALU,
// memory-bound once on WMMA; total WMMA work is only 2x useful (256/129).
// All staging fits in the 192 MB L2, so pass-2 fragment loads hit cache.
// ---------------------------------------------------------------------------

typedef __attribute__((ext_vector_type(16))) _Float16 v16h;
typedef __attribute__((ext_vector_type(8)))  _Float16 v8h;
typedef __attribute__((ext_vector_type(8)))  float    v8f;

#define BB   32
#define TT   8192
#define CC   64
#define NP   2048          // BB*CC pairs
#define NL   129           // 2*MD+1 lags
#define APAD 8352          // a_pad row length (valid data at [64, 8256))
#define BPAD 8576          // b_pad row length (valid data at [240, 8432))
#define TSWEEP 8320        // 260 k-steps of 32

// --------------------------- kernel 1: zero halos ---------------------------
// a halo: [0,64) + [8256,8352) = 160 elems ; b halo: [0,240)+[8432,8576)=384.
__global__ void __launch_bounds__(256) corr_zero_pads(_Float16* ah, _Float16* bh) {
    int gid = blockIdx.x * 256 + threadIdx.x;       // exactly NP*544 threads
    int p = gid / 544;
    int r = gid - p * 544;
    if (r < 64)        ah[(size_t)p * APAD + r] = (_Float16)0.0f;
    else if (r < 160)  ah[(size_t)p * APAD + 8256 + (r - 64)] = (_Float16)0.0f;
    else if (r < 400)  bh[(size_t)p * BPAD + (r - 160)] = (_Float16)0.0f;
    else               bh[(size_t)p * BPAD + 8432 + (r - 400)] = (_Float16)0.0f;
}

// ------------- kernel 2: sumsq partials + f16 convert + transpose -----------
// grid: blockIdx.x = b*16 + chunk (512 blocks, 256 threads).
// Reads [b][t][c] fp32 coalesced; LDS-transposes 64x64 tiles (stride-68 pad,
// conflict-free) so writes to [pair][t] f16 layout are contiguous 32B runs.
__global__ void __launch_bounds__(256)
corr_pass1(const float* __restrict__ in1, const float* __restrict__ in2,
           _Float16* __restrict__ ah, _Float16* __restrict__ bh,
           float* __restrict__ s1p, float* __restrict__ s2p) {
    __shared__ _Float16 la[64 * 68];
    __shared__ _Float16 lb[64 * 68];
    __shared__ float red[256];

    const int b     = blockIdx.x >> 4;
    const int chunk = blockIdx.x & 15;
    const int tid   = threadIdx.x;
    const int c     = tid & 63;
    const int tq    = tid >> 6;           // 0..3
    const int tbase = chunk * 512;

    float acc1 = 0.0f, acc2 = 0.0f;

    for (int tile = 0; tile < 8; ++tile) {
        const int tt = tbase + tile * 64;
        #pragma unroll 4
        for (int sub = 0; sub < 16; ++sub) {
            const int tl = sub * 4 + tq;
            const size_t gi = ((size_t)(b * TT + (tt + tl))) * CC + c;
            float v1 = in1[gi];
            float v2 = in2[gi];
            acc1 += v1 * v1;
            acc2 += v2 * v2;
            la[c * 68 + tl] = (_Float16)v1;   // transpose into LDS
            lb[c * 68 + tl] = (_Float16)v2;
        }
        __syncthreads();
        // write-out: 4 threads per channel row, 16 halves (32B) each
        {
            const int r   = tid >> 2;       // channel 0..63
            const int seg = tid & 3;
            const size_t po = (size_t)(b * CC + r);
            _Float16* pa = ah + po * APAD + 64  + tt + seg * 16;
            _Float16* pb = bh + po * BPAD + 240 + tt + seg * 16;
            const _Float16* sa = &la[r * 68 + seg * 16];
            const _Float16* sb = &lb[r * 68 + seg * 16];
            #pragma unroll
            for (int i = 0; i < 16; ++i) pa[i] = sa[i];
            #pragma unroll
            for (int i = 0; i < 16; ++i) pb[i] = sb[i];
        }
        __syncthreads();
    }

    // per-channel reduce of the 4 tq-partials (deterministic; no float atomics)
    red[tid] = acc1;
    __syncthreads();
    if (tq == 0)
        s1p[chunk * NP + b * CC + c] = red[c] + red[64 + c] + red[128 + c] + red[192 + c];
    __syncthreads();
    red[tid] = acc2;
    __syncthreads();
    if (tq == 0)
        s2p[chunk * NP + b * CC + c] = red[c] + red[64 + c] + red[128 + c] + red[192 + c];
}

// --------------- kernel 3: fixed-order reduction of partials ----------------
__global__ void __launch_bounds__(256)
corr_reduce_norms(const float* __restrict__ s1p, const float* __restrict__ s2p,
                  float* __restrict__ s1, float* __restrict__ s2) {
    int i = blockIdx.x * 256 + threadIdx.x;   // 0..4095
    const float* src = (i < NP) ? s1p : s2p;
    float*       dst = (i < NP) ? s1  : s2;
    int p = i & (NP - 1);
    float s = 0.0f;
    #pragma unroll
    for (int ch = 0; ch < 16; ++ch) s += src[ch * NP + p];
    dst[p] = s;
}

// ----------------- kernel 4: Toeplitz WMMA correlation ----------------------
// 512 blocks x 128 threads = 2048 wave32 waves; wave <-> (b,c) pair.
// Inner loop: 8x global_load_b128 + 2x v_wmma_f32_16x16x32_f16 per 64 k-steps
// slice (branch-free; compiler software-pipelines with s_wait_loadcnt).
__global__ void __launch_bounds__(128)
corr_wmma(const _Float16* __restrict__ ah, const _Float16* __restrict__ bh,
          const float* __restrict__ s1, const float* __restrict__ s2,
          float* __restrict__ out) {
    const int lane = threadIdx.x & 31;
    const int wave = threadIdx.x >> 5;
    const int p    = blockIdx.x * 4 + wave;   // pair id 0..2047
    const int b    = p >> 6;
    const int c    = p & 63;

    const _Float16* arow = ah + (size_t)p * APAD;
    const _Float16* brow = bh + (size_t)p * BPAD;

    // A layout (matches B): lanes 0-15 = rows M=0..15 holding K=0..15;
    // lanes 16-31 = rows M=0..15 holding K=16..31.
    const int m    = lane & 15;
    const int aoff = m + ((lane < 16) ? 0 : 16);
    // B layout: lane 0-15 = column n holding K=0..15; lanes 16-31: K=16..31.
    const int n    = lane & 15;
    const int boff = ((lane < 16) ? 0 : 16) - 16 * n + 240;   // >= 0, 32B aligned

    // Warm the cache path once (global_prefetch_b8); data is L2-resident,
    // so no in-loop prefetch -> keeps the inner loop branch-free.
    __builtin_prefetch(arow + aoff, 0, 1);
    __builtin_prefetch(brow + boff, 0, 1);

    // Scalar norm loads issued early; consumed after the loop.
    const float s1v = s1[p];
    const float s2v = s2[p];

    v8f acc = {0.f, 0.f, 0.f, 0.f, 0.f, 0.f, 0.f, 0.f};

    #pragma unroll 4
    for (int t0 = 0; t0 < TSWEEP; t0 += 32) {
        // A fragment: two 16B loads per lane, 2B-aligned (sliding window).
        v8h a0, a1;
        __builtin_memcpy(&a0, arow + t0 + aoff,      16);
        __builtin_memcpy(&a1, arow + t0 + aoff + 8, 16);
        v16h af;
        #pragma unroll
        for (int i = 0; i < 8; ++i) { af[i] = a0[i]; af[i + 8] = a1[i]; }

        // B fragment: 32B contiguous per lane, 32B aligned -> direct b128 loads.
        const v8h* pb = (const v8h*)(brow + t0 + boff);
        v8h q0 = pb[0], q1 = pb[1];
        v16h bf;
        #pragma unroll
        for (int i = 0; i < 8; ++i) { bf[i] = q0[i]; bf[i + 8] = q1[i]; }

        acc = __builtin_amdgcn_wmma_f32_16x16x32_f16(
                  /*neg_a=*/false, af, /*neg_b=*/false, bf,
                  /*c_mod=*/(short)0, acc, /*reuse_a=*/false, /*reuse_b=*/false);
    }

    // deferred L2 normalization (scalar per (b,c) commutes with correlation)
    const float sc = rsqrtf(fmaxf(s1v, 1e-12f)) * rsqrtf(fmaxf(s2v, 1e-12f));

    // C/D layout: lane 0-15 -> N=lane, VGPR v -> M=v; lanes 16-31 -> M=v+8.
    const int mb = (lane < 16) ? 0 : 8;
    #pragma unroll
    for (int v = 0; v < 8; ++v) {
        const int lag = mb + v + 16 * n;       // bijective over 0..255
        if (lag < NL)
            out[((size_t)b * NL + lag) * CC + c] = acc[v] * sc;
    }
}

// ------------------------------- launcher -----------------------------------
extern "C" void kernel_launch(void* const* d_in, const int* in_sizes, int n_in,
                              void* d_out, int out_size, void* d_ws, size_t ws_size,
                              hipStream_t stream) {
    (void)in_sizes; (void)n_in; (void)out_size; (void)ws_size;
    const float* in1 = (const float*)d_in[0];
    const float* in2 = (const float*)d_in[1];
    float* out = (float*)d_out;

    char* ws = (char*)d_ws;
    const size_t szA  = (size_t)NP * APAD * sizeof(_Float16);   // 34,209,792 B
    const size_t szB  = (size_t)NP * BPAD * sizeof(_Float16);   // 35,127,296 B
    const size_t szSP = (size_t)16 * NP * sizeof(float);        //    131,072 B
    _Float16* ah  = (_Float16*)(ws);
    _Float16* bh  = (_Float16*)(ws + szA);
    float*    s1p = (float*)(ws + szA + szB);
    float*    s2p = (float*)(ws + szA + szB + szSP);
    float*    s1  = (float*)(ws + szA + szB + 2 * szSP);
    float*    s2  = (float*)(ws + szA + szB + 2 * szSP + NP * sizeof(float));
    // total ws use ~69.6 MB

    corr_zero_pads<<<(NP * 544) / 256, 256, 0, stream>>>(ah, bh);
    corr_pass1<<<BB * 16, 256, 0, stream>>>(in1, in2, ah, bh, s1p, s2p);
    corr_reduce_norms<<<16, 256, 0, stream>>>(s1p, s2p, s1, s2);
    corr_wmma<<<NP / 4, 128, 0, stream>>>(ah, bh, s1, s2, out);
}
